// CrossAttention_60782377173156
// MI455X (gfx1250) — matrix operand order repaired and checked by reading, not compiled, not run on hardware
//
#include <hip/hip_runtime.h>
#include <hip/hip_bf16.h>

typedef __attribute__((ext_vector_type(16))) _Float16 v16h;
typedef __attribute__((ext_vector_type(8)))  _Float16 v8h;
typedef __attribute__((ext_vector_type(8)))  float    v8f;
typedef __attribute__((ext_vector_type(4)))  int      v4i;

#define BATCH 4
#define CTOT  768
#define HD    64      // head dim == CQ
#define CKV   128
#define SEQ   4096    // 64*64
#define IMGW  64

#if __has_builtin(__builtin_amdgcn_global_load_async_to_lds_b128) && \
    __has_builtin(__builtin_amdgcn_s_wait_asynccnt)
#define HAVE_ASYNC_LDS 1
#endif

__device__ __forceinline__ v8f wmma_f16(v16h a, v16h b, v8f c) {
  // D = A(16x32 f16) * B(32x16 f16) + C(16x16 f32)
  return __builtin_amdgcn_wmma_f32_16x16x32_f16(false, a, false, b, (short)0, c, false, false);
}

__device__ __forceinline__ v16h cat16(v8h lo, v8h hi) {
  return __builtin_shufflevector(lo, hi, 0, 1, 2, 3, 4, 5, 6, 7,
                                 8, 9, 10, 11, 12, 13, 14, 15);
}

// B-operand fragment: per-lane K = half*16 + e (16 contiguous halves at p).
__device__ __forceinline__ v16h load_bfrag(const _Float16* p) {
  return cat16(*(const v8h*)p, *(const v8h*)(p + 8));
}
// A-operand fragment: per-lane K = half*16 + e (16 contiguous halves at p,
// p already includes half*16).
__device__ __forceinline__ v16h load_afrag(const _Float16* p) {
  return cat16(*(const v8h*)p, *(const v8h*)(p + 8));
}

#ifdef HAVE_ASYNC_LDS
typedef __attribute__((address_space(1))) v4i gv4i_t;  // global int4
typedef __attribute__((address_space(3))) v4i sv4i_t;  // LDS int4
__device__ __forceinline__ void async_g2l_b128(const _Float16* g, _Float16* l) {
  __builtin_amdgcn_global_load_async_to_lds_b128(
      (gv4i_t*)(void*)const_cast<_Float16*>(g),
      (sv4i_t*)(void*)l, 0, 0);
}
#endif

// ---------------------------------------------------------------------------
__global__ void zero_f32(float* p, int n) {
  int i = blockIdx.x * blockDim.x + threadIdx.x;
  if (i < n) p[i] = 0.f;
}

__global__ void f32_to_f16(const float* __restrict__ s, _Float16* __restrict__ d, int n) {
  int i = blockIdx.x * blockDim.x + threadIdx.x;
  if (i < n) d[i] = (_Float16)s[i];
}

// ---------------------------------------------------------------------------
// grouped 1x1 projections
// ---------------------------------------------------------------------------
// q stored TRANSPOSED: q_t[(b*SEQ + p)*HD + g]  (seq-major, for A-frag loads)
__global__ void proj_q_kernel(const float* __restrict__ x,
                              const float* __restrict__ wq,
                              const float* __restrict__ bq,
                              _Float16* __restrict__ q_t) {
  int idx = blockIdx.x * blockDim.x + threadIdx.x;      // B*64*SEQ
  if (idx >= BATCH * HD * SEQ) return;
  int p = idx % SEQ;
  int g = (idx / SEQ) % HD;
  int b = idx / (SEQ * HD);
  const float* xb = x + ((size_t)b * CTOT + (size_t)g * 12) * SEQ + p;
  float acc = bq[g];
#pragma unroll
  for (int i = 0; i < 12; ++i) acc += xb[(size_t)i * SEQ] * wq[g * 12 + i];
  q_t[((size_t)b * SEQ + p) * HD + g] = (_Float16)(acc * 0.125f); // fold 1/sqrt(64)
}

__global__ void proj_kv_kernel(const float* __restrict__ xe,
                               const float* __restrict__ wkv,
                               const float* __restrict__ bkv,
                               float* __restrict__ kv) {
  int idx = blockIdx.x * blockDim.x + threadIdx.x;      // B*128*SEQ
  if (idx >= BATCH * CKV * SEQ) return;
  int p = idx % SEQ;
  int g = (idx / SEQ) % CKV;
  int b = idx / (SEQ * CKV);
  const float* xb = xe + ((size_t)b * CTOT + (size_t)g * 6) * SEQ + p;
  float acc = bkv[g];
#pragma unroll
  for (int i = 0; i < 6; ++i) acc += xb[(size_t)i * SEQ] * wkv[g * 6 + i];
  kv[idx] = acc;
}

// ---------------------------------------------------------------------------
// conv branch (KSxKS, 128->128) as implicit GEMM with WMMA.
// out ch 0..63  -> K, stored seq-major:  k_tr[(b*SEQ+pos)*HD + oc]   (f16)
// out ch 64..127-> V, stored ch-major:   v_h[((b*HD)+oc-64)*SEQ+pos] (f16)
// grid: B * 2 * (SEQ/16), block: 128 (4 waves, one 16-out-ch tile each)
// ---------------------------------------------------------------------------
template <int KS>
__global__ void conv_branch_kernel(const float* __restrict__ kv,
                                   const _Float16* __restrict__ wh,  // (128, RED) f16
                                   const float* __restrict__ bias,
                                   _Float16* __restrict__ k_tr,
                                   _Float16* __restrict__ v_h) {
  constexpr int KK  = KS * KS;
  constexpr int PAD = KS / 2;
  constexpr int RED = CKV * KK;                 // divisible by 32 for 3/5/7
  __shared__ __align__(16) _Float16 Bs[16 * 32];  // [pixel][red-chunk]

  int bid  = blockIdx.x;
  int npix = bid % (SEQ / 16);  bid /= (SEQ / 16);
  int mh   = bid % 2;           bid /= 2;
  int b    = bid;
  int wave = threadIdx.x / 32;
  int lane = threadIdx.x % 32;
  int hl   = lane / 16, ln = lane % 16;
  int mbase   = mh * 64 + wave * 16;
  int pixbase = npix * 16;
  const float* inb = kv + (size_t)b * CKV * SEQ;
  const _Float16* wrow = wh + (size_t)(mbase + ln) * RED + hl * 16;

  v8f acc = {};
  for (int kb = 0; kb < RED; kb += 32) {
    __syncthreads();
    // cooperative im2col staging: Bs[p][r], 512 f16
    for (int t = threadIdx.x; t < 512; t += blockDim.x) {
      int p   = pixbase + t / 32;
      int r   = kb + (t % 32);
      int cin = r / KK;
      int rr  = r % KK;
      int kh  = rr / KS, kw = rr % KS;
      int y   = p / IMGW, x = p % IMGW;
      int iy  = y + kh - PAD, ix = x + kw - PAD;
      float v = 0.f;
      if (iy >= 0 && iy < IMGW && ix >= 0 && ix < IMGW)
        v = inb[(size_t)cin * SEQ + (size_t)iy * IMGW + ix];
      Bs[t] = (_Float16)v;
    }
    __syncthreads();
    v16h af = load_afrag(wrow + kb);               // 2x global_load_b128
    v16h bf = load_bfrag(&Bs[ln * 32 + hl * 16]);  // 2x ds_load_b128
    acc = wmma_f16(af, bf, acc);
  }
  // D layout: M = j + 8*hl (out ch), N = ln (pixel)
  size_t pos = (size_t)pixbase + ln;
  if (mh == 0) {
    v8h hv;
#pragma unroll
    for (int j = 0; j < 8; ++j) hv[j] = (_Float16)(acc[j] + bias[mbase + j + 8 * hl]);
    // 8 contiguous channels -> one 16B store
    *(v8h*)&k_tr[((size_t)b * SEQ + pos) * HD + mbase + 8 * hl] = hv;
  } else {
#pragma unroll
    for (int j = 0; j < 8; ++j) {
      int oc = mbase + j + 8 * hl;           // 64..127
      v_h[((size_t)b * HD + (oc - HD)) * SEQ + pos] =
          (_Float16)(acc[j] + bias[oc]);
    }
  }
}

// ---------------------------------------------------------------------------
// fused flash attention: O += softmax(Q^T K) V^T   (scale folded into Q)
// q_t: (b, seq, ch) f16; k_tr: (b, seq, ch) f16; v_h: (b, ch, seq) f16
// grid: B * (SEQ/64), block: 128 (4 waves, 16 queries each)
// ---------------------------------------------------------------------------
__global__ void flash_attn_kernel(const _Float16* __restrict__ q_t,
                                  const _Float16* __restrict__ k_tr,
                                  const _Float16* __restrict__ v_h,
                                  float* __restrict__ o_acc) {
  __shared__ __align__(16) _Float16 Kty[32 * HD];     // [key][ch]
  __shared__ __align__(16) _Float16 Vty[HD * 32];     // [ch][key]
  __shared__ __align__(16) _Float16 Pbuf[4][16 * 32]; // per-wave [q][key]

  int b     = blockIdx.x / (SEQ / 64);
  int qblk  = (blockIdx.x % (SEQ / 64)) * 64;
  int wave  = threadIdx.x / 32;
  int lane  = threadIdx.x % 32;
  int hl    = lane / 16, ln = lane % 16;
  int qbase = qblk + wave * 16;

  const _Float16* qb = q_t + (size_t)b * SEQ * HD;
  const _Float16* kb = k_tr + (size_t)b * SEQ * HD;
  const _Float16* vb = v_h + (size_t)b * HD * SEQ;

  // Q as A-operand, two 32-ch chunks: A[m=ln][k=ch]
  const _Float16* qrow = qb + ((size_t)(qbase + ln)) * HD + hl * 16;
  v16h aq0 = load_afrag(qrow);
  v16h aq1 = load_afrag(qrow + 32);

  v8f oacc[4] = {};
  float mrow[8], lrow[8];
#pragma unroll
  for (int j = 0; j < 8; ++j) { mrow[j] = -1e30f; lrow[j] = 0.f; }

  for (int k0 = 0; k0 < SEQ; k0 += 32) {
    __syncthreads();
#ifdef HAVE_ASYNC_LDS
    {
      // async direct global->LDS staging (ASYNCcnt path, bypasses VGPRs)
      int t = threadIdx.x;
      const _Float16* ksrc = kb + (size_t)k0 * HD;          // contiguous 4KB
      async_g2l_b128(ksrc + (size_t)t * 8, Kty + t * 8);
      async_g2l_b128(ksrc + (size_t)(t + 128) * 8, Kty + (t + 128) * 8);
      int ch = t / 2, part = t % 2;                          // V: 64 rows x 32
      const _Float16* vsrc = vb + (size_t)ch * SEQ + k0 + part * 16;
      _Float16* vdst = Vty + ch * 32 + part * 16;
      async_g2l_b128(vsrc, vdst);
      async_g2l_b128(vsrc + 8, vdst + 8);
      __builtin_amdgcn_s_wait_asynccnt(0);
    }
#else
    {
      const uint4* src = (const uint4*)(kb + (size_t)k0 * HD);
      uint4* dst = (uint4*)Kty;
      for (int t = threadIdx.x; t < (32 * HD) / 8; t += blockDim.x)
        dst[t] = src[t];
      int t = threadIdx.x;
      int ch = t / 2, part = t % 2;
      const uint4* vs = (const uint4*)(vb + (size_t)ch * SEQ + k0 + part * 16);
      uint4* vd = (uint4*)(Vty + ch * 32 + part * 16);
      vd[0] = vs[0];
      vd[1] = vs[1];
    }
#endif
    __syncthreads();

    // S tiles (16q x 32keys): B[k=ch][n=key] -> Kty[key][ch] contiguous in ch
    v8f s0 = {}, s1 = {};
    {
      const _Float16* kr0 = &Kty[ln * HD];          // key = ln
      const _Float16* kr1 = &Kty[(16 + ln) * HD];   // key = 16+ln
      s0 = wmma_f16(aq0, load_bfrag(kr0 + hl * 16), s0);
      s1 = wmma_f16(aq0, load_bfrag(kr1 + hl * 16), s1);
      s0 = wmma_f16(aq1, load_bfrag(kr0 + 32 + hl * 16), s0);
      s1 = wmma_f16(aq1, load_bfrag(kr1 + 32 + hl * 16), s1);
    }

    // ---- online softmax; row (j + 8*hl) lives across 16 lanes of a half ----
    // stage-major butterflies: 8 independent bpermutes per stage -> 1 wait
    float tmax[8];
#pragma unroll
    for (int j = 0; j < 8; ++j) tmax[j] = fmaxf(s0[j], s1[j]);
#pragma unroll
    for (int m = 1; m < 16; m <<= 1) {
      float u[8];
#pragma unroll
      for (int j = 0; j < 8; ++j) u[j] = __shfl_xor(tmax[j], m, 32);
#pragma unroll
      for (int j = 0; j < 8; ++j) tmax[j] = fmaxf(tmax[j], u[j]);
    }
    float corr[8], rs[8];
#pragma unroll
    for (int j = 0; j < 8; ++j) {
      float mn = fmaxf(mrow[j], tmax[j]);
      corr[j]  = __expf(mrow[j] - mn);
      mrow[j]  = mn;
      float p0 = __expf(s0[j] - mn);
      float p1 = __expf(s1[j] - mn);
      s0[j] = p0; s1[j] = p1;
      rs[j] = p0 + p1;
    }
#pragma unroll
    for (int m = 1; m < 16; m <<= 1) {
      float u[8];
#pragma unroll
      for (int j = 0; j < 8; ++j) u[j] = __shfl_xor(rs[j], m, 32);
#pragma unroll
      for (int j = 0; j < 8; ++j) rs[j] += u[j];
    }
#pragma unroll
    for (int j = 0; j < 8; ++j) {
      lrow[j] = lrow[j] * corr[j] + rs[j];
      oacc[0][j] *= corr[j]; oacc[1][j] *= corr[j];
      oacc[2][j] *= corr[j]; oacc[3][j] *= corr[j];
    }

    // P (C/D layout) -> A-operand layout via per-wave LDS
    _Float16* P = &Pbuf[wave][0];
#pragma unroll
    for (int j = 0; j < 8; ++j) {
      int m = j + 8 * hl;
      P[m * 32 + ln]      = (_Float16)s0[j];
      P[m * 32 + 16 + ln] = (_Float16)s1[j];
    }
    v16h ap = load_afrag(&P[ln * 32 + hl * 16]);    // 2x ds_load_b128

    // O += P(16x32) * V^T(32keys x 16ch): B[k=key][n=ch] -> Vty[ch][key]
#pragma unroll
    for (int cht = 0; cht < 4; ++cht) {
      v16h bv = load_bfrag(&Vty[(cht * 16 + ln) * 32 + hl * 16]);
      oacc[cht] = wmma_f16(ap, bv, oacc[cht]);
    }
  }

  // normalize; accumulate branch sum into o_acc (b, ch, seq)
#pragma unroll
  for (int cht = 0; cht < 4; ++cht) {
#pragma unroll
    for (int j = 0; j < 8; ++j) {
      float inv = 1.0f / lrow[j];
      int ch = cht * 16 + ln;
      int q  = qbase + j + 8 * hl;
      float* dst = o_acc + ((size_t)b * HD + ch) * SEQ + q;
      *dst += oacc[cht][j] * inv;
    }
  }
}

// ---------------------------------------------------------------------------
// output conv 3x3 (64 -> 768) + bias + residual (implicit GEMM WMMA)
// grid: B * (768/64) * (SEQ/16), block: 128
// ---------------------------------------------------------------------------
__global__ void out_conv_kernel(const float* __restrict__ attn,      // (b,64,seq)
                                const _Float16* __restrict__ wh,     // (768, 576) f16
                                const float* __restrict__ bias,
                                const float* __restrict__ xres,
                                float* __restrict__ out) {
  constexpr int KS = 3, KK = 9, PAD = 1;
  constexpr int RED = HD * KK;                  // 576 = 18*32
  __shared__ __align__(16) _Float16 Bs[16 * 32];

  int bid  = blockIdx.x;
  int npix = bid % (SEQ / 16);  bid /= (SEQ / 16);
  int mt   = bid % (CTOT / 64); bid /= (CTOT / 64);
  int b    = bid;
  int wave = threadIdx.x / 32;
  int lane = threadIdx.x % 32;
  int hl   = lane / 16, ln = lane % 16;
  int mbase   = mt * 64 + wave * 16;
  int pixbase = npix * 16;
  const float* inb = attn + (size_t)b * HD * SEQ;
  const _Float16* wrow = wh + (size_t)(mbase + ln) * RED + hl * 16;

  v8f acc = {};
  for (int kb2 = 0; kb2 < RED; kb2 += 32) {
    __syncthreads();
    for (int t = threadIdx.x; t < 512; t += blockDim.x) {
      int p   = pixbase + t / 32;
      int r   = kb2 + (t % 32);
      int cin = r / KK;
      int rr  = r % KK;
      int kh  = rr / KS, kw = rr % KS;
      int y   = p / IMGW, x = p % IMGW;
      int iy  = y + kh - PAD, ix = x + kw - PAD;
      float v = 0.f;
      if (iy >= 0 && iy < IMGW && ix >= 0 && ix < IMGW)
        v = inb[(size_t)cin * SEQ + (size_t)iy * IMGW + ix];
      Bs[t] = (_Float16)v;
    }
    __syncthreads();
    v16h af = load_afrag(wrow + kb2);
    v16h bf = load_bfrag(&Bs[ln * 32 + hl * 16]);
    acc = wmma_f16(af, bf, acc);
  }
#pragma unroll
  for (int j = 0; j < 8; ++j) {
    int oc = mbase + j + 8 * hl;
    size_t gi = ((size_t)b * CTOT + oc) * SEQ + pixbase + ln;
    out[gi] = acc[j] + bias[oc] + xres[gi];
  }
}

// ---------------------------------------------------------------------------
extern "C" void kernel_launch(void* const* d_in, const int* in_sizes, int n_in,
                              void* d_out, int out_size, void* d_ws, size_t ws_size,
                              hipStream_t stream) {
  const float* x    = (const float*)d_in[0];
  const float* xe   = (const float*)d_in[1];
  const float* wq   = (const float*)d_in[2];
  const float* bq   = (const float*)d_in[3];
  const float* wkv  = (const float*)d_in[4];
  const float* bkv  = (const float*)d_in[5];
  const float* w3   = (const float*)d_in[6];
  const float* b3   = (const float*)d_in[7];
  const float* w5   = (const float*)d_in[8];
  const float* b5   = (const float*)d_in[9];
  const float* w7   = (const float*)d_in[10];
  const float* b7   = (const float*)d_in[11];
  const float* wout = (const float*)d_in[12];
  const float* bout = (const float*)d_in[13];
  float* out = (float*)d_out;

  char* ws = (char*)d_ws;
  size_t off = 0;
  auto take = [&](size_t bytes) {
    char* p = ws + off;
    off = (off + bytes + 255) & ~(size_t)255;
    return (void*)p;
  };
  _Float16* q_t  = (_Float16*)take((size_t)BATCH * SEQ * HD * 2);   // 2 MB
  float*    kv_f = (float*)   take((size_t)BATCH * CKV * SEQ * 4);  // 8 MB
  _Float16* k_tr = (_Float16*)take((size_t)BATCH * SEQ * HD * 2);   // 2 MB
  _Float16* v_h  = (_Float16*)take((size_t)BATCH * HD * SEQ * 2);   // 2 MB
  float*    oatt = (float*)   take((size_t)BATCH * HD * SEQ * 4);   // 4 MB
  _Float16* wh3  = (_Float16*)take((size_t)CKV * CKV * 9 * 2);
  _Float16* wh5  = (_Float16*)take((size_t)CKV * CKV * 25 * 2);
  _Float16* wh7  = (_Float16*)take((size_t)CKV * CKV * 49 * 2);
  _Float16* whO  = (_Float16*)take((size_t)CTOT * HD * 9 * 2);
  (void)ws_size;

  const int nq  = BATCH * HD * SEQ;
  const int nkv = BATCH * CKV * SEQ;

  zero_f32<<<(nq + 255) / 256, 256, 0, stream>>>(oatt, nq);
  proj_q_kernel<<<(nq + 255) / 256, 256, 0, stream>>>(x, wq, bq, q_t);
  proj_kv_kernel<<<(nkv + 255) / 256, 256, 0, stream>>>(xe, wkv, bkv, kv_f);

  const int n3 = CKV * CKV * 9, n5 = CKV * CKV * 25, n7 = CKV * CKV * 49;
  const int nO = CTOT * HD * 9;
  f32_to_f16<<<(n3 + 255) / 256, 256, 0, stream>>>(w3, wh3, n3);
  f32_to_f16<<<(n5 + 255) / 256, 256, 0, stream>>>(w5, wh5, n5);
  f32_to_f16<<<(n7 + 255) / 256, 256, 0, stream>>>(w7, wh7, n7);
  f32_to_f16<<<(nO + 255) / 256, 256, 0, stream>>>(wout, whO, nO);

  dim3 convGrid(BATCH * 2 * (SEQ / 16));
  dim3 attnGrid(BATCH * (SEQ / 64));

  conv_branch_kernel<3><<<convGrid, 128, 0, stream>>>(kv_f, wh3, b3, k_tr, v_h);
  flash_attn_kernel<<<attnGrid, 128, 0, stream>>>(q_t, k_tr, v_h, oatt);

  conv_branch_kernel<5><<<convGrid, 128, 0, stream>>>(kv_f, wh5, b5, k_tr, v_h);
  flash_attn_kernel<<<attnGrid, 128, 0, stream>>>(q_t, k_tr, v_h, oatt);

  conv_branch_kernel<7><<<convGrid, 128, 0, stream>>>(kv_f, wh7, b7, k_tr, v_h);
  flash_attn_kernel<<<attnGrid, 128, 0, stream>>>(q_t, k_tr, v_h, oatt);

  out_conv_kernel<<<dim3(BATCH * (CTOT / 64) * (SEQ / 16)), 128, 0, stream>>>(
      oatt, whO, bout, x, out);
}
